// MoE_29850022708069
// MI455X (gfx1250) — matrix operand order repaired and checked by reading, not compiled, not run on hardware
//
#include <hip/hip_runtime.h>
#include <hip/hip_bf16.h>

// ---------------------------------------------------------------------------
// MoE dense SwiGLU for MI455X (gfx1250), wave32 + v_wmma_f32_16x16x32_bf16.
//
// GEMM operands are pre-converted (fp32->bf16) and pre-permuted into WMMA
// fragment layouts (ISA 7.12.2): [tile][kblock][lane][8 dwords]. GEMM inner
// loops are contiguous per-lane 32B global loads (2x global_load_b128 per
// operand) + v_wmma, software-pipelined via loadcnt; no LDS, no barriers.
// Waves compute multi-tile register blocks for operand reuse:
//   ffn1: 32x32 act tile  -> 8 wmma / 6KB loads per k-block (21.3 FLOP/B)
//   ffn2: 32x32 out tile  -> 4 wmma / 4KB loads per k-block (16   FLOP/B)
//
//  A (16x32 bf16) elem(m,k): lane = m + 16*((k%16)/8), vgpr = 4*(k/16)+(k%8)/2
//  B (32x16 bf16) elem(k,n): lane = n + 16*((k%16)/8), vgpr = 4*(k/16)+(k%8)/2
//
// ffn2 is the algebraically-fused single GEMM: out = act' @ Wout_flat with
// K = E*FF = 16384 (gate weight folded into act' in ffn1's epilogue).
// ---------------------------------------------------------------------------

typedef __attribute__((ext_vector_type(16))) __bf16 v16bf;
typedef __attribute__((ext_vector_type(8)))  float  v8f;

#define S_TOK  2048
#define H_DIM  1024
#define FF_DIM 1024
#define E_NUM  16
#define CLIPV  7.0f

static __device__ __forceinline__ unsigned short f32_bf16(float f) {
    unsigned int u = __builtin_bit_cast(unsigned int, f);
    u += 0x7FFFu + ((u >> 16) & 1u);           // round-to-nearest-even
    return (unsigned short)(u >> 16);
}

static __device__ __forceinline__ v8f wmma_bf16(v16bf a, v16bf b, v8f c) {
    return __builtin_amdgcn_wmma_f32_16x16x32_bf16(false, a, false, b,
                                                   (short)0, c, false, false);
}

// ---------------------------------------------------------------------------
// fp32 row-major [S][K]  ->  bf16 A-fragment blocks [S/16][K/32][32 lanes][8]
// ---------------------------------------------------------------------------
__global__ void k_to_afrag(const float* __restrict__ src,
                           unsigned int* __restrict__ dst, int K) {
    int stile = blockIdx.x, kb = blockIdx.y;
    int tid = threadIdx.x;
    unsigned int* out = dst + ((size_t)stile * gridDim.y + kb) * 256;
    const float* base = src + ((size_t)stile * 16) * K + kb * 32;
#pragma unroll
    for (int i = 0; i < 2; ++i) {
        int idx  = tid + i * 128;            // 0..255 == lane*8 + vgpr
        int lane = idx >> 3, vgpr = idx & 7;
        int m = lane & 15, hb = lane >> 4;
        int k = (vgpr >> 2) * 16 + hb * 8 + (vgpr & 3) * 2;
        const float* p = base + (size_t)m * K + k;
        unsigned lo = f32_bf16(p[0]), hi = f32_bf16(p[1]);
        out[idx] = lo | (hi << 16);
    }
}

// ---------------------------------------------------------------------------
// fp32 row-major [K][N] -> bf16 B-fragment blocks [N/16][K/32][32 lanes][8]
// grid = (N/16, K/32, nMat)
// ---------------------------------------------------------------------------
__global__ void k_to_bfrag(const float* __restrict__ src,
                           unsigned int* __restrict__ dst, int K, int N) {
    int dt = blockIdx.x, kb = blockIdx.y, z = blockIdx.z;
    int tid = threadIdx.x;
    size_t perz = (size_t)gridDim.x * gridDim.y * 256;
    unsigned int* out = dst + (size_t)z * perz +
                        ((size_t)dt * gridDim.y + kb) * 256;
    const float* base = src + (size_t)z * K * N + (size_t)(kb * 32) * N + dt * 16;
#pragma unroll
    for (int i = 0; i < 2; ++i) {
        int idx  = tid + i * 128;
        int lane = idx >> 3, vgpr = idx & 7;
        int n = lane & 15, khi = lane >> 4;
        int k = (vgpr >> 2) * 16 + khi * 8 + (vgpr & 3) * 2;
        const float* p = base + (size_t)k * N + n;
        unsigned lo = f32_bf16(p[0]), hi = f32_bf16(p[N]);
        out[idx] = lo | (hi << 16);
    }
}

__global__ void k_zero_i32(int* __restrict__ p, int n) {
    int i = threadIdx.x;
    if (i < n) p[i] = 0;
}

// ---------------------------------------------------------------------------
// Router (fp32, exact top-k semantics). Block = 256 = 16 tokens x 16 experts.
// ---------------------------------------------------------------------------
__global__ void k_router(const float* __restrict__ x, const float* __restrict__ rw,
                         const float* __restrict__ rb, float* __restrict__ fw,
                         float* __restrict__ probs, int* __restrict__ load_cnt) {
    __shared__ float lg[16][17];
    int tid = threadIdx.x;
    int e = tid & 15, tl = tid >> 4;
    int tok = blockIdx.x * 16 + tl;

    const float4* xr = (const float4*)(x  + (size_t)tok * H_DIM);
    const float4* wr = (const float4*)(rw + (size_t)e   * H_DIM);
    float acc = rb[e];
    for (int k = 0; k < H_DIM / 4; ++k) {
        float4 a = xr[k], b = wr[k];
        acc += a.x * b.x + a.y * b.y + a.z * b.z + a.w * b.w;
    }
    lg[tl][e] = acc;
    __syncthreads();

    if (tid < 16) {
        int t = tid;
        tok = blockIdx.x * 16 + t;
        float v[16];
        for (int j = 0; j < 16; ++j) v[j] = lg[t][j];

        float M = v[0];
        for (int j = 1; j < 16; ++j) M = fmaxf(M, v[j]);
        float p[16], se = 0.f;
        for (int j = 0; j < 16; ++j) { p[j] = __expf(v[j] - M); se += p[j]; }
        float inv = 1.f / se;
        for (int j = 0; j < 16; ++j) probs[(size_t)tok * 16 + j] = p[j] * inv;

        int ti[4]; float tv[4]; unsigned used = 0;
        for (int kk = 0; kk < 4; ++kk) {
            float best = -3.4e38f; int bi = 0;
            for (int j = 0; j < 16; ++j)
                if (!((used >> j) & 1) && v[j] > best) { best = v[j]; bi = j; }
            used |= 1u << bi; ti[kk] = bi; tv[kk] = best;
        }
        float w4[4], s4 = 0.f;
        for (int kk = 0; kk < 4; ++kk) { w4[kk] = __expf(tv[kk] - tv[0]); s4 += w4[kk]; }
        float in4 = 1.f / s4;

        float o16[16];
        for (int j = 0; j < 16; ++j) o16[j] = 0.f;
        for (int kk = 0; kk < 4; ++kk) o16[ti[kk]] += w4[kk] * in4;
        for (int j = 0; j < 16; ++j) fw[(size_t)tok * 16 + j] = o16[j];

        int pos = 0;
        for (int kk = 1; kk < 4; ++kk) if (ti[kk] > ti[pos]) pos = kk;
        atomicAdd(&load_cnt[pos], 1);
    }
}

// be[s,h] = sum_e fw[s,e]*b_out[e,h]   (grid = (H/256, S))
__global__ void k_biaseff(const float* __restrict__ fw, const float* __restrict__ b_out,
                          float* __restrict__ be) {
    int h = blockIdx.x * 256 + threadIdx.x;
    int s = blockIdx.y;
    float acc = 0.f;
#pragma unroll
    for (int e2 = 0; e2 < 16; ++e2)
        acc += fw[(size_t)s * 16 + e2] * b_out[(size_t)e2 * H_DIM + h];
    be[(size_t)s * H_DIM + h] = acc;
}

// ---------------------------------------------------------------------------
// ffn1: per expert e, act'[s, e*FF+f] = w[s,e]*silu(clip(gate))*clip(up).
// Block = 128 (4 waves, 2x2). Wave tile: 32 rows x 32 act cols
//  => 8 wmma per k-block (2 A frags x (2 up + 2 gate) B frags).
// Output written directly in ffn2's A-fragment order.
// grid = (FF/64, S/64, E)
// ---------------------------------------------------------------------------
__global__ void __launch_bounds__(128)
k_ffn1(const unsigned int* __restrict__ xaf, const unsigned int* __restrict__ wif,
       const float* __restrict__ b_in, const float* __restrict__ fw,
       unsigned short* __restrict__ actf) {
    int tid = threadIdx.x, lane = tid & 31, w = tid >> 5;
    int rwv = w >> 1, cwv = w & 1;
    int e = blockIdx.z;
    int stile0 = blockIdx.y * 4 + rwv * 2;
    int fbase  = blockIdx.x * 64 + cwv * 32;   // act-col base for this wave
    int ft0 = fbase >> 4;                      // 16-col tile idx (0..63)
    const int KB = H_DIM / 32;                 // 32 k-blocks

    const unsigned int* A0  = xaf + ((size_t)stile0 * KB) * 256 + lane * 8;
    const unsigned int* A1  = A0 + (size_t)KB * 256;
    const unsigned int* Bu0 = wif + (((size_t)e * 128 + ft0) * KB) * 256 + lane * 8;
    const unsigned int* Bu1 = Bu0 + (size_t)KB * 256;
    const unsigned int* Bg0 = Bu0 + (size_t)64 * KB * 256;
    const unsigned int* Bg1 = Bg0 + (size_t)KB * 256;

    v8f z8 = {0, 0, 0, 0, 0, 0, 0, 0};
    v8f cu[2][2] = {{z8, z8}, {z8, z8}};
    v8f cg[2][2] = {{z8, z8}, {z8, z8}};

#pragma unroll 2
    for (int kb = 0; kb < KB; ++kb) {
        __builtin_prefetch(A0  + (kb + 8) * 256, 0, 1);   // global_prefetch_b8
        __builtin_prefetch(A1  + (kb + 8) * 256, 0, 1);
        __builtin_prefetch(Bu0 + (kb + 8) * 256, 0, 1);
        __builtin_prefetch(Bu1 + (kb + 8) * 256, 0, 1);
        __builtin_prefetch(Bg0 + (kb + 8) * 256, 0, 1);
        __builtin_prefetch(Bg1 + (kb + 8) * 256, 0, 1);
        v16bf a0  = *(const v16bf*)(A0  + kb * 256);
        v16bf a1  = *(const v16bf*)(A1  + kb * 256);
        v16bf bu0 = *(const v16bf*)(Bu0 + kb * 256);
        v16bf bu1 = *(const v16bf*)(Bu1 + kb * 256);
        v16bf bg0 = *(const v16bf*)(Bg0 + kb * 256);
        v16bf bg1 = *(const v16bf*)(Bg1 + kb * 256);
        cu[0][0] = wmma_bf16(a0, bu0, cu[0][0]);
        cu[0][1] = wmma_bf16(a0, bu1, cu[0][1]);
        cg[0][0] = wmma_bf16(a0, bg0, cg[0][0]);
        cg[0][1] = wmma_bf16(a0, bg1, cg[0][1]);
        cu[1][0] = wmma_bf16(a1, bu0, cu[1][0]);
        cu[1][1] = wmma_bf16(a1, bu1, cu[1][1]);
        cg[1][0] = wmma_bf16(a1, bg0, cg[1][0]);
        cg[1][1] = wmma_bf16(a1, bg1, cg[1][1]);
    }

    // epilogue: bias + clip + silu, scale by gate weight, store bf16 directly
    // into ffn2's A-fragment layout [stile][kbg][lane'][vgpr'][half]
    int n = lane & 15, mh = (lane >> 4) * 8;
#pragma unroll
    for (int it = 0; it < 2; ++it) {
        int stile = stile0 + it;
#pragma unroll
        for (int jt = 0; jt < 2; ++jt) {
            int f0 = fbase + jt * 16;
            float bub = b_in[(size_t)e * (2 * FF_DIM) + f0 + n];
            float bgb = b_in[(size_t)e * (2 * FF_DIM) + FF_DIM + f0 + n];
            int kbg   = e * 32 + (f0 >> 5);
            int vgprp = ((f0 & 16) >> 2) + ((n & 7) >> 1);
            int lanehp = 16 * (n >> 3);
            int halfb = (n & 1) * 2;
            size_t blockbase = (((size_t)stile * 512 + kbg) * 32) * 32;  // bytes
#pragma unroll
            for (int r = 0; r < 8; ++r) {
                int m = mh + r;
                int s = stile * 16 + m;
                float u = fminf(fmaxf(cu[it][jt][r] + bub, -CLIPV), CLIPV);
                float g = fminf(fmaxf(cg[it][jt][r] + bgb, -CLIPV), CLIPV);
                float act = (g / (1.f + __expf(-g))) * u;
                float scale = fw[(size_t)s * 16 + e];
                *(unsigned short*)((char*)actf + blockbase +
                                   (size_t)(lanehp + m) * 32 + vgprp * 4 + halfb) =
                    f32_bf16(act * scale);
            }
        }
    }
}

// ---------------------------------------------------------------------------
// ffn2: out[s,h] = act'[s,:] @ Wout_flat[:,h] + be[s,h]
// Single dense GEMM, K = E*FF = 16384. Block = 128 (4 waves, 2x2).
// Wave tile: 32 rows x 32 cols => 4 wmma per k-block. grid = (H/64, S/64)
// ---------------------------------------------------------------------------
__global__ void __launch_bounds__(128)
k_ffn2(const unsigned int* __restrict__ actf, const unsigned int* __restrict__ wof,
       const float* __restrict__ be, float* __restrict__ out) {
    int tid = threadIdx.x, lane = tid & 31, w = tid >> 5;
    int rwv = w >> 1, cwv = w & 1;
    int stile0 = blockIdx.y * 4 + rwv * 2;
    int dtb = blockIdx.x * 4 + cwv * 2;            // two 16-col tiles
    const int KB = (E_NUM * FF_DIM) / 32;          // 512 k-blocks

    const unsigned int* A0 = actf + ((size_t)stile0 * KB) * 256 + lane * 8;
    const unsigned int* A1 = A0 + (size_t)KB * 256;
    const unsigned int* B0 = wof + ((size_t)dtb * KB) * 256 + lane * 8;
    const unsigned int* B1 = B0 + (size_t)KB * 256;

    v8f z8 = {0, 0, 0, 0, 0, 0, 0, 0};
    v8f c[2][2] = {{z8, z8}, {z8, z8}};

#pragma unroll 2
    for (int kb = 0; kb < KB; ++kb) {
        __builtin_prefetch(A0 + (kb + 8) * 256, 0, 1);
        __builtin_prefetch(A1 + (kb + 8) * 256, 0, 1);
        __builtin_prefetch(B0 + (kb + 8) * 256, 0, 1);
        __builtin_prefetch(B1 + (kb + 8) * 256, 0, 1);
        v16bf a0 = *(const v16bf*)(A0 + kb * 256);
        v16bf a1 = *(const v16bf*)(A1 + kb * 256);
        v16bf b0 = *(const v16bf*)(B0 + kb * 256);
        v16bf b1 = *(const v16bf*)(B1 + kb * 256);
        c[0][0] = wmma_bf16(a0, b0, c[0][0]);
        c[0][1] = wmma_bf16(a0, b1, c[0][1]);
        c[1][0] = wmma_bf16(a1, b0, c[1][0]);
        c[1][1] = wmma_bf16(a1, b1, c[1][1]);
    }

    int n = lane & 15, mh = (lane >> 4) * 8;
#pragma unroll
    for (int it = 0; it < 2; ++it) {
        int stile = stile0 + it;
#pragma unroll
        for (int jt = 0; jt < 2; ++jt) {
            int h = (dtb + jt) * 16 + n;
#pragma unroll
            for (int r = 0; r < 8; ++r) {
                int s = stile * 16 + mh + r;
                out[(size_t)s * H_DIM + h] = c[it][jt][r] + be[(size_t)s * H_DIM + h];
            }
        }
    }
}

// ---------------------------------------------------------------------------
// aux = 0.02 * E * sum_e importance[e] * load[e]   (deterministic column sums)
// ---------------------------------------------------------------------------
__global__ void k_finalize(const float* __restrict__ probs,
                           const int* __restrict__ cnt, float* __restrict__ aux) {
    __shared__ float imp[16];
    int e = threadIdx.x;
    if (e < 16) {
        float s = 0.f;
        for (int t = 0; t < S_TOK; ++t) s += probs[(size_t)t * 16 + e];
        imp[e] = s * (1.f / S_TOK);
    }
    __syncthreads();
    if (e == 0) {
        float a = 0.f;
        for (int j = 0; j < 16; ++j)
            a += imp[j] * ((float)cnt[j] * (1.f / S_TOK));
        *aux = 0.02f * 16.f * a;
    }
}

// ---------------------------------------------------------------------------
extern "C" void kernel_launch(void* const* d_in, const int* in_sizes, int n_in,
                              void* d_out, int out_size, void* d_ws, size_t ws_size,
                              hipStream_t stream) {
    (void)in_sizes; (void)n_in; (void)out_size; (void)ws_size;
    const float* x    = (const float*)d_in[0];
    const float* Win  = (const float*)d_in[1];
    const float* bin  = (const float*)d_in[2];
    const float* Wout = (const float*)d_in[3];
    const float* bout = (const float*)d_in[4];
    const float* rw   = (const float*)d_in[5];
    const float* rb   = (const float*)d_in[6];
    float* out = (float*)d_out;

    // workspace layout (bytes)
    char* ws = (char*)d_ws;
    unsigned int* xaf = (unsigned int*)(ws);                                  // 4 MiB
    unsigned int* wif = (unsigned int*)(ws + (size_t)4   * 1024 * 1024);      // 64 MiB
    unsigned int* wof = (unsigned int*)(ws + (size_t)68  * 1024 * 1024);      // 32 MiB
    unsigned int* acf = (unsigned int*)(ws + (size_t)100 * 1024 * 1024);      // 64 MiB
    float* fw    = (float*)(ws + (size_t)164 * 1024 * 1024);                  // 128 KiB
    float* probs = fw + (size_t)S_TOK * E_NUM;                                // 128 KiB
    int*   cnt   = (int*)(probs + (size_t)S_TOK * E_NUM);                     // 64 B
    float* be    = (float*)(ws + (size_t)165 * 1024 * 1024);                  // 8 MiB

    k_zero_i32<<<1, 32, 0, stream>>>(cnt, 16);

    dim3 gx(S_TOK / 16, H_DIM / 32);
    k_to_afrag<<<gx, 128, 0, stream>>>(x, xaf, H_DIM);
    dim3 gwi((2 * FF_DIM) / 16, H_DIM / 32, E_NUM);
    k_to_bfrag<<<gwi, 128, 0, stream>>>(Win, wif, H_DIM, 2 * FF_DIM);
    dim3 gwo(H_DIM / 16, (E_NUM * FF_DIM) / 32, 1);
    k_to_bfrag<<<gwo, 128, 0, stream>>>(Wout, wof, E_NUM * FF_DIM, H_DIM);

    k_router<<<S_TOK / 16, 256, 0, stream>>>(x, rw, rb, fw, probs, cnt);

    dim3 gb(H_DIM / 256, S_TOK);
    k_biaseff<<<gb, 256, 0, stream>>>(fw, bout, be);

    dim3 g1(FF_DIM / 64, S_TOK / 64, E_NUM);
    k_ffn1<<<g1, 128, 0, stream>>>(xaf, wif, bin, fw, (unsigned short*)acf);

    dim3 g2(H_DIM / 64, S_TOK / 64, 1);
    k_ffn2<<<g2, 128, 0, stream>>>(acf, wof, be, out);

    k_finalize<<<1, 16, 0, stream>>>(probs, cnt, out + (size_t)S_TOK * H_DIM);
}
